// DeformableTransformerDecoderLayer_59811714564392
// MI455X (gfx1250) — matrix operand order repaired and checked by reading, not compiled, not run on hardware
//
#include <hip/hip_runtime.h>
#include <hip/hip_bf16.h>
#include <math.h>

// ---------------------------------------------------------------------------
// Problem constants (from the reference): E=256, NH=8, HD=32, NL=4, NP=4,
// DFFN=1024, nq=900, bs=8, Ls=100, Lv=21760. Level starts hardcoded from
// SHAPES = [[128,128],[64,64],[32,32],[16,16]].
// ---------------------------------------------------------------------------

typedef __attribute__((ext_vector_type(16))) _Float16 v16h;
typedef __attribute__((ext_vector_type(8)))  _Float16 v8h;
typedef __attribute__((ext_vector_type(2)))  _Float16 v2h;
typedef __attribute__((ext_vector_type(8)))  float    v8f;

#define GEMM_BM 128
#define GEMM_BN 64
#define GEMM_KB 32
#define LDH 40           // halfs per LDS row: 80B pitch -> 16B-aligned frags,
                         // lane windows tile all 64 banks (m*20 mod 64 distinct)

// A fragment per ISA 16-bit A 16x32 layout:
// lanes 0-15 (M=lane):   f[0..7]=K0..7,  f[8..15]=K16..23
// lanes 16-31(M=lane-16):f[0..7]=K8..15, f[8..15]=K24..31
__device__ inline v16h load_frag_A(const _Float16* lds, int rowBase, int lane) {
    int m  = lane & 15;
    int ko = (lane >> 4) * 8;
    const _Float16* p = lds + (rowBase + m) * LDH + ko;
    v8h a = *reinterpret_cast<const v8h*>(p);        // ds_load_b128
    v8h b = *reinterpret_cast<const v8h*>(p + 16);   // ds_load_b128
    return __builtin_shufflevector(a, b, 0, 1, 2, 3, 4, 5, 6, 7,
                                   8, 9, 10, 11, 12, 13, 14, 15);
}

// B fragment per ISA 16-bit B 32x16 layout (from B 64x16 pattern):
// lanes 0-15 (N=lane):   f[0..7]=K0..7,  f[8..15]=K16..23
// lanes 16-31(N=lane-16):f[0..7]=K8..15, f[8..15]=K24..31
__device__ inline v16h load_frag_B(const _Float16* lds, int rowBase, int lane) {
    int n  = lane & 15;
    int ko = (lane >> 4) * 8;
    const _Float16* p = lds + (rowBase + n) * LDH + ko;
    v8h a = *reinterpret_cast<const v8h*>(p);
    v8h b = *reinterpret_cast<const v8h*>(p + 16);
    return __builtin_shufflevector(a, b, 0, 1, 2, 3, 4, 5, 6, 7,
                                   8, 9, 10, 11, 12, 13, 14, 15);
}

// Stage one K-step of A (BM x 32) and B (BN x 32) tiles as f16 hi/lo.
// GUARD=false -> block fully interior in M: unconditional b64 global loads.
template <bool GUARD>
__device__ inline void gemm_stage(const float* __restrict__ A,
                                  const float* __restrict__ W,
                                  int M, int K, int m0, int n0, int k0, int tid,
                                  _Float16 Ah[][LDH], _Float16 Al[][LDH],
                                  _Float16 Bh[][LDH], _Float16 Bl[][LDH])
{
#pragma unroll
    for (int i = 0; i < 8; ++i) {            // 2048 float2 / 256 thr = 8
        int e = i * 256 + tid;               // pair index
        int r = e >> 4, c = (e & 15) * 2;
        int gr = m0 + r;
        float2 v = {0.f, 0.f};
        if (!GUARD || gr < M)
            v = *reinterpret_cast<const float2*>(&A[(size_t)gr * K + k0 + c]);
        _Float16 h0 = (_Float16)v.x, h1 = (_Float16)v.y;
        v2h hp = {h0, h1};
        v2h lp = {(_Float16)(v.x - (float)h0), (_Float16)(v.y - (float)h1)};
        *reinterpret_cast<v2h*>(&Ah[r][c]) = hp;   // ds_store_b32
        *reinterpret_cast<v2h*>(&Al[r][c]) = lp;
    }
#pragma unroll
    for (int i = 0; i < 4; ++i) {            // 1024 float2 / 256 thr = 4
        int e = i * 256 + tid;
        int r = e >> 4, c = (e & 15) * 2;
        float2 v = *reinterpret_cast<const float2*>(
            &W[(size_t)(n0 + r) * K + k0 + c]);
        _Float16 h0 = (_Float16)v.x, h1 = (_Float16)v.y;
        v2h hp = {h0, h1};
        v2h lp = {(_Float16)(v.x - (float)h0), (_Float16)(v.y - (float)h1)};
        *reinterpret_cast<v2h*>(&Bh[r][c]) = hp;
        *reinterpret_cast<v2h*>(&Bl[r][c]) = lp;
    }
}

// C = act(A[M,K] @ W[N,K]^T + bias[N]); f32 in/out, split f16 hi/lo WMMA core.
// Requires K % 32 == 0, N % 64 == 0. M arbitrary (guarded edge blocks).
__global__ __launch_bounds__(256) void gemm_wmma_kernel(
    const float* __restrict__ A, const float* __restrict__ W,
    const float* __restrict__ bias, float* __restrict__ C,
    int M, int N, int K, int act)
{
    __shared__ __align__(16) _Float16 Ah[GEMM_BM][LDH];
    __shared__ __align__(16) _Float16 Al[GEMM_BM][LDH];
    __shared__ __align__(16) _Float16 Bh[GEMM_BN][LDH];
    __shared__ __align__(16) _Float16 Bl[GEMM_BN][LDH];

    const int tid  = threadIdx.x;
    const int lane = tid & 31;
    const int wave = tid >> 5;
    const int wrow = wave & 3;   // 4 wave-rows -> 128 M
    const int wcol = wave >> 2;  // 2 wave-cols -> 64  N
    const int m0 = blockIdx.y * GEMM_BM;
    const int n0 = blockIdx.x * GEMM_BN;
    const bool edge = (m0 + GEMM_BM > M);    // uniform across block

    v8f acc[2][2] = {};

    for (int k0 = 0; k0 < K; k0 += GEMM_KB) {
        if (edge)
            gemm_stage<true >(A, W, M, K, m0, n0, k0, tid, Ah, Al, Bh, Bl);
        else
            gemm_stage<false>(A, W, M, K, m0, n0, k0, tid, Ah, Al, Bh, Bl);

        // Prefetch next k-tile of A (speculative; emits global_prefetch_b8).
        if (k0 + GEMM_KB < K) {
            int gr = m0 + (tid >> 1);
            if (!edge || gr < M)
                __builtin_prefetch(&A[(size_t)gr * K + k0 + GEMM_KB], 0, 1);
        }
        __syncthreads();

        v16h ah[2], al[2], bh[2], bl[2];
#pragma unroll
        for (int i = 0; i < 2; ++i) {
            ah[i] = load_frag_A(&Ah[0][0], wrow * 32 + i * 16, lane);
            al[i] = load_frag_A(&Al[0][0], wrow * 32 + i * 16, lane);
            bh[i] = load_frag_B(&Bh[0][0], wcol * 32 + i * 16, lane);
            bl[i] = load_frag_B(&Bl[0][0], wcol * 32 + i * 16, lane);
        }
#pragma unroll
        for (int mi = 0; mi < 2; ++mi)
#pragma unroll
            for (int ni = 0; ni < 2; ++ni) {
                acc[mi][ni] = __builtin_amdgcn_wmma_f32_16x16x32_f16(
                    false, ah[mi], false, bh[ni], (short)0, acc[mi][ni], false, false);
                acc[mi][ni] = __builtin_amdgcn_wmma_f32_16x16x32_f16(
                    false, al[mi], false, bh[ni], (short)0, acc[mi][ni], false, false);
                acc[mi][ni] = __builtin_amdgcn_wmma_f32_16x16x32_f16(
                    false, ah[mi], false, bl[ni], (short)0, acc[mi][ni], false, false);
            }
        __syncthreads();
    }

    // Store per f32 C layout: lane n = lane&15, rows r + 8*(lane>>4).
    const int lmo = (lane >> 4) * 8;
    const int ln  = lane & 15;
    if (!edge) {
#pragma unroll
        for (int mi = 0; mi < 2; ++mi)
#pragma unroll
            for (int ni = 0; ni < 2; ++ni) {
                int n = n0 + wcol * 32 + ni * 16 + ln;
                float bv = bias ? bias[n] : 0.f;
#pragma unroll
                for (int r = 0; r < 8; ++r) {
                    int m = m0 + wrow * 32 + mi * 16 + lmo + r;
                    float v = acc[mi][ni][r] + bv;
                    if (act == 1) v = v > 0.f ? v : 0.f;
                    C[(size_t)m * N + n] = v;
                }
            }
    } else {
#pragma unroll
        for (int mi = 0; mi < 2; ++mi)
#pragma unroll
            for (int ni = 0; ni < 2; ++ni) {
                int n = n0 + wcol * 32 + ni * 16 + ln;
                float bv = bias ? bias[n] : 0.f;
#pragma unroll
                for (int r = 0; r < 8; ++r) {
                    int m = m0 + wrow * 32 + mi * 16 + lmo + r;
                    if (m < M) {
                        float v = acc[mi][ni][r] + bv;
                        if (act == 1) v = v > 0.f ? v : 0.f;
                        C[(size_t)m * N + n] = v;
                    }
                }
            }
    }
}

// ---------------------------------------------------------------------------
// Flash-style MHA core (VALU). One thread per query; K/V tiles staged in LDS.
// Element addressing: X[l * ls + b * bs + h*32 + d].
// ---------------------------------------------------------------------------
#define ATT_QT 256
#define ATT_KT 32

__global__ __launch_bounds__(256) void attention_kernel(
    const float* __restrict__ Q, int q_ls, int q_bs,
    const float* __restrict__ Kp, int k_ls, int k_bs,
    const float* __restrict__ Vp, int v_ls, int v_bs,
    float* __restrict__ O, int o_ls, int o_bs,
    int Lq, int Lk, float scale)
{
    __shared__ float Ks[ATT_KT][33];
    __shared__ float Vs[ATT_KT][33];
    const int bh = blockIdx.y;
    const int b = bh >> 3;      // NH = 8
    const int h = bh & 7;
    const int q = blockIdx.x * ATT_QT + threadIdx.x;
    const bool active = q < Lq;

    float qv[32];
    const float* qptr = Q + (size_t)(active ? q : 0) * q_ls + (size_t)b * q_bs + h * 32;
#pragma unroll
    for (int d = 0; d < 32; ++d) qv[d] = active ? qptr[d] : 0.f;

    float ov[32] = {};
    float mx = -1e30f, lsum = 0.f;

    for (int kk0 = 0; kk0 < Lk; kk0 += ATT_KT) {
        for (int e = threadIdx.x; e < ATT_KT * 32; e += 256) {
            int r = e >> 5, c = e & 31;
            int kidx = kk0 + r;
            float kvv = 0.f, vvv = 0.f;
            if (kidx < Lk) {
                kvv = Kp[(size_t)kidx * k_ls + (size_t)b * k_bs + h * 32 + c];
                vvv = Vp[(size_t)kidx * v_ls + (size_t)b * v_bs + h * 32 + c];
            }
            Ks[r][c] = kvv;
            Vs[r][c] = vvv;
        }
        __syncthreads();

        float s[ATT_KT];
        float tmax = -1e30f;
#pragma unroll
        for (int r = 0; r < ATT_KT; ++r) {
            float dacc = 0.f;
#pragma unroll
            for (int c = 0; c < 32; ++c) dacc += qv[c] * Ks[r][c];
            dacc *= scale;
            if (kk0 + r >= Lk) dacc = -1e30f;
            s[r] = dacc;
            tmax = fmaxf(tmax, dacc);
        }
        float nm = fmaxf(mx, tmax);
        float corr = __expf(mx - nm);
        mx = nm;
        lsum *= corr;
#pragma unroll
        for (int d = 0; d < 32; ++d) ov[d] *= corr;
#pragma unroll
        for (int r = 0; r < ATT_KT; ++r) {
            float p = __expf(s[r] - nm);
            lsum += p;
#pragma unroll
            for (int c = 0; c < 32; ++c) ov[c] += p * Vs[r][c];
        }
        __syncthreads();
    }

    if (active) {
        float inv = 1.f / lsum;
        float* optr = O + (size_t)q * o_ls + (size_t)b * o_bs + h * 32;
#pragma unroll
        for (int d = 0; d < 32; ++d) optr[d] = ov[d] * inv;
    }
}

// ---------------------------------------------------------------------------
// out[row] = LN(X[row] + R[row]) * w + b ; E = 256, one block per row.
// ---------------------------------------------------------------------------
__global__ __launch_bounds__(256) void add_ln_kernel(
    const float* __restrict__ X, const float* __restrict__ R,
    const float* __restrict__ w, const float* __restrict__ bias,
    float* __restrict__ out)
{
    __shared__ float sred[8];
    __shared__ float sstat;
    const size_t row = blockIdx.x;
    const int tid = threadIdx.x;
    float x = X[row * 256 + tid] + R[row * 256 + tid];

    float sum = x;
#pragma unroll
    for (int o = 16; o > 0; o >>= 1) sum += __shfl_down(sum, o, 32);
    if ((tid & 31) == 0) sred[tid >> 5] = sum;
    __syncthreads();
    if (tid == 0) {
        float s = 0.f;
#pragma unroll
        for (int i = 0; i < 8; ++i) s += sred[i];
        sstat = s * (1.f / 256.f);
    }
    __syncthreads();
    const float mean = sstat;
    float d = x - mean;
    float sq = d * d;
#pragma unroll
    for (int o = 16; o > 0; o >>= 1) sq += __shfl_down(sq, o, 32);
    __syncthreads();
    if ((tid & 31) == 0) sred[tid >> 5] = sq;
    __syncthreads();
    if (tid == 0) {
        float s = 0.f;
#pragma unroll
        for (int i = 0; i < 8; ++i) s += sred[i];
        sstat = s * (1.f / 256.f);
    }
    __syncthreads();
    out[row * 256 + tid] = d * rsqrtf(sstat + 1e-5f) * w[tid] + bias[tid];
}

__global__ __launch_bounds__(256) void add_kernel(
    const float* __restrict__ a, const float* __restrict__ b,
    float* __restrict__ o, int n)
{
    int i = blockIdx.x * 256 + threadIdx.x;
    if (i < n) o[i] = a[i] + b[i];
}

// ---------------------------------------------------------------------------
// Deformable attention sampling. One block per (q*8+b); thread = (h, d).
// val rows are (lv*8 + b), cols h*32+d. off/aw rows are (q*8 + b).
// ---------------------------------------------------------------------------
__device__ inline float dsample(const float* __restrict__ val, int base,
                                int H, int W, int y, int x, int b, int col)
{
    if (x < 0 || x >= W || y < 0 || y >= H) return 0.f;
    size_t row = ((size_t)(base + y * W + x) * 8 + (size_t)b);
    return val[row * 256 + col];
}

__global__ __launch_bounds__(256) void deform_kernel(
    const float* __restrict__ val,
    const float* __restrict__ off_raw,  // [(q*8+b), 256] (NH,NL,NP,2)
    const float* __restrict__ aw_raw,   // [(q*8+b), 128] (NH,NL,NP)
    const float* __restrict__ ref,      // [nq, bs, NL, 4]
    float* __restrict__ out)            // [(q*8+b), 256]
{
    __shared__ float loc[256];
    __shared__ float wsm[128];
    const int bq = blockIdx.x;       // q*8 + b
    const int b = bq & 7;
    const int tid = threadIdx.x;

    {   // loc = xy + off/NP * wh * 0.5  (NP=4 -> factor 0.125)
        float offv = off_raw[(size_t)bq * 256 + tid];
        int c = tid & 1;
        int l = (tid >> 3) & 3;
        size_t rbase = ((size_t)bq * 4 + l) * 4;  // ((q*8+b)*NL + l)*4
        loc[tid] = ref[rbase + c] + offv * 0.125f * ref[rbase + 2 + c];
    }
    if (tid < 128) wsm[tid] = aw_raw[(size_t)bq * 128 + tid];
    __syncthreads();
    if (tid < 8) {  // softmax over NL*NP=16 per head
        float mxv = -1e30f;
        for (int i = 0; i < 16; ++i) mxv = fmaxf(mxv, wsm[tid * 16 + i]);
        float s = 0.f;
        for (int i = 0; i < 16; ++i) {
            float e = __expf(wsm[tid * 16 + i] - mxv);
            wsm[tid * 16 + i] = e;
            s += e;
        }
        float inv = 1.f / s;
        for (int i = 0; i < 16; ++i) wsm[tid * 16 + i] *= inv;
    }
    __syncthreads();

    const int h = tid >> 5, d = tid & 31;
    const int col = h * 32 + d;
    const int Hs[4]     = {128, 64, 32, 16};
    const int starts[4] = {0, 16384, 20480, 21504};
    float acc = 0.f;
#pragma unroll
    for (int l = 0; l < 4; ++l) {
        const int H = Hs[l], W = Hs[l], base = starts[l];
#pragma unroll
        for (int p = 0; p < 4; ++p) {
            int li = ((h * 4 + l) * 4 + p);
            float x = loc[li * 2 + 0] * (float)W - 0.5f;
            float y = loc[li * 2 + 1] * (float)H - 0.5f;
            float x0f = floorf(x), y0f = floorf(y);
            float lx = x - x0f, ly = y - y0f;
            int x0 = (int)x0f, y0 = (int)y0f;
            float s00 = dsample(val, base, H, W, y0,     x0,     b, col);
            float s01 = dsample(val, base, H, W, y0,     x0 + 1, b, col);
            float s10 = dsample(val, base, H, W, y0 + 1, x0,     b, col);
            float s11 = dsample(val, base, H, W, y0 + 1, x0 + 1, b, col);
            float bi = s00 * (1.f - lx) * (1.f - ly) + s01 * lx * (1.f - ly)
                     + s10 * (1.f - lx) * ly         + s11 * lx * ly;
            acc += wsm[li] * bi;
        }
    }
    out[(size_t)bq * 256 + tid] = acc;
}

// ---------------------------------------------------------------------------
// Host orchestration
// ---------------------------------------------------------------------------
static inline void launch_gemm(const float* A, const float* W, const float* bias,
                               float* C, int M, int N, int K, int act,
                               hipStream_t s)
{
    dim3 g(N / GEMM_BN, (M + GEMM_BM - 1) / GEMM_BM);
    gemm_wmma_kernel<<<g, 256, 0, s>>>(A, W, bias, C, M, N, K, act);
}

extern "C" void kernel_launch(void* const* d_in, const int* in_sizes, int n_in,
                              void* d_out, int out_size, void* d_ws, size_t ws_size,
                              hipStream_t stream)
{
    const float* tgt      = (const float*)d_in[0];
    const float* tgt_pos  = (const float*)d_in[1];
    const float* ref_pts  = (const float*)d_in[2];
    const float* mem_sup  = (const float*)d_in[3];   // [8,100,256]
    const float* memory   = (const float*)d_in[4];   // [21760,8,256]
    const float* sa_in_w  = (const float*)d_in[6];
    const float* sa_in_b  = (const float*)d_in[7];
    const float* sa_out_w = (const float*)d_in[8];
    const float* sa_out_b = (const float*)d_in[9];
    const float* cs_in_w  = (const float*)d_in[10];
    const float* cs_in_b  = (const float*)d_in[11];
    const float* cs_out_w = (const float*)d_in[12];
    const float* cs_out_b = (const float*)d_in[13];
    const float* so_w     = (const float*)d_in[14];
    const float* so_b     = (const float*)d_in[15];
    const float* aw_w     = (const float*)d_in[16];
    const float* aw_b     = (const float*)d_in[17];
    const float* vp_w     = (const float*)d_in[18];
    const float* vp_b     = (const float*)d_in[19];
    const float* op_w     = (const float*)d_in[20];
    const float* op_b     = (const float*)d_in[21];
    const float* norm1_w  = (const float*)d_in[22];
    const float* norm1_b  = (const float*)d_in[23];
    const float* norm2_w  = (const float*)d_in[24];
    const float* norm2_b  = (const float*)d_in[25];
    const float* csn_w    = (const float*)d_in[26];
    const float* csn_b    = (const float*)d_in[27];
    const float* norm3_w  = (const float*)d_in[28];
    const float* norm3_b  = (const float*)d_in[29];
    const float* lin1_w   = (const float*)d_in[30];
    const float* lin1_b   = (const float*)d_in[31];
    const float* lin2_w   = (const float*)d_in[32];
    const float* lin2_b   = (const float*)d_in[33];

    const int ROWS = 7200;              // nq*bs
    const int MVAL = 174080;            // Lv*bs
    const float scale = 0.17677669529663689f;  // 1/sqrt(32)

    float* ws = (float*)d_ws;
    size_t off = 0;
    float* q_buf = ws + off; off += 1843200;   // q1 / q2 / query
    float* qk    = ws + off; off += 3686400;   // sa qk / (qh2 + kv2) / aw_raw
    float* vbuf  = ws + off; off += 1843200;   // sa v / off_raw
    float* attn  = ws + off; off += 1843200;   // attn outs / deform out / ffn2
    float* proj  = ws + off; off += 1843200;   // pre-LN projection
    float* tgtA  = ws + off; off += 1843200;
    float* tgtB  = ws + off; off += 1843200;
    float* val   = ws + off; off += 44564480;  // vp output; FFN hidden aliases
    float* kv2   = qk + 1843200;               // [800,512]
    float* ffn1  = val;                        // [7200,1024] aliases dead val

    dim3 b256(256);

    // ---- Self-attention -------------------------------------------------
    add_kernel<<<ROWS, b256, 0, stream>>>(tgt, tgt_pos, q_buf, ROWS * 256);
    launch_gemm(q_buf, sa_in_w, sa_in_b, qk, ROWS, 512, 256, 0, stream);
    launch_gemm(tgt, sa_in_w + 512 * 256, sa_in_b + 512, vbuf, ROWS, 256, 256, 0, stream);
    {
        dim3 g((900 + ATT_QT - 1) / ATT_QT, 64);
        attention_kernel<<<g, b256, 0, stream>>>(
            qk,        4096, 512,        // Q rows (l*8+b), stride 512
            qk + 256,  4096, 512,        // K
            vbuf,      2048, 256,        // V
            attn,      2048, 256,
            900, 900, scale);
    }
    launch_gemm(attn, sa_out_w, sa_out_b, proj, ROWS, 256, 256, 0, stream);
    add_ln_kernel<<<ROWS, b256, 0, stream>>>(tgt, proj, norm2_w, norm2_b, tgtA);

    // ---- Cross-attention on memory_support ------------------------------
    add_kernel<<<ROWS, b256, 0, stream>>>(tgtA, tgt_pos, q_buf, ROWS * 256);
    launch_gemm(q_buf, cs_in_w, cs_in_b, qk, ROWS, 256, 256, 0, stream);
    launch_gemm(mem_sup, cs_in_w + 256 * 256, cs_in_b + 256, kv2, 800, 512, 256, 0, stream);
    {
        dim3 g((900 + ATT_QT - 1) / ATT_QT, 64);
        attention_kernel<<<g, b256, 0, stream>>>(
            qk,        2048, 256,        // Q [7200,256]
            kv2,       512,  51200,      // K rows (b*100+l), stride 512
            kv2 + 256, 512,  51200,      // V
            attn,      2048, 256,
            900, 100, scale);
    }
    launch_gemm(attn, cs_out_w, cs_out_b, proj, ROWS, 256, 256, 0, stream);
    add_ln_kernel<<<ROWS, b256, 0, stream>>>(tgtA, proj, csn_w, csn_b, tgtB);

    // ---- Deformable attention ------------------------------------------
    add_kernel<<<ROWS, b256, 0, stream>>>(tgtB, tgt_pos, q_buf, ROWS * 256);
    launch_gemm(q_buf, so_w, so_b, vbuf, ROWS, 256, 256, 0, stream);   // offsets
    launch_gemm(q_buf, aw_w, aw_b, qk,   ROWS, 128, 256, 0, stream);   // attn weights
    launch_gemm(memory, vp_w, vp_b, val, MVAL, 256, 256, 0, stream);   // value proj
    deform_kernel<<<ROWS, b256, 0, stream>>>(val, vbuf, qk, ref_pts, attn);
    launch_gemm(attn, op_w, op_b, proj, ROWS, 256, 256, 0, stream);
    add_ln_kernel<<<ROWS, b256, 0, stream>>>(tgtB, proj, norm1_w, norm1_b, tgtA);

    // ---- FFN ------------------------------------------------------------
    launch_gemm(tgtA, lin1_w, lin1_b, ffn1, ROWS, 1024, 256, 1, stream);  // relu
    launch_gemm(ffn1, lin2_w, lin2_b, attn, ROWS, 256, 1024, 0, stream);
    add_ln_kernel<<<ROWS, b256, 0, stream>>>(tgtA, attn, norm3_w, norm3_b,
                                             (float*)d_out);
}
